// KANLayer_26225070309752
// MI455X (gfx1250) — matrix operand order repaired and checked by reading, not compiled, not run on hardware
//
#include <hip/hip_runtime.h>
#include <hip/hip_bf16.h>

// ---------------------------------------------------------------------------
// KAN layer forward for MI455X (gfx1250, wave32, WMMA).
//   acts(4096x4608) fp16  = [8 cubic B-spline bases + silu] * x   (in d_ws)
//   out (4096x512)  fp32  = acts @ weight^T + bias   via v_wmma_f32_16x16x32_f16
// GEMM: 128x64 tile / 4 wave32s, 64x32 per wave (8 accumulators), BK=32,
//       double-buffered LDS, one barrier per K-step. Uniform base pointers
//       advance by BK (SALU); per-thread 32-bit offsets keep VGPR pressure low
//       (round-3 pointer arrays caused scratch spills).
// ---------------------------------------------------------------------------

typedef __attribute__((ext_vector_type(16))) _Float16 v16h;
typedef __attribute__((ext_vector_type(8)))  _Float16 v8h;
typedef __attribute__((ext_vector_type(4)))  _Float16 v4h;
typedef __attribute__((ext_vector_type(8)))  float    v8f;

#define BATCH   4096
#define IN_DIM  512
#define OUT_DIM 512
#define NBASIS  9                   // NUM + K + 1 = 5 + 3 + 1
#define KDIM    (IN_DIM * NBASIS)   // 4608

#define BM 128
#define BN 64
#define BK 32
#define NITER (KDIM / BK)           // 144

// ---------------------------------------------------------------------------
// Kernel 1: weight fp32 -> fp16 (same (o, k) row-major layout, k contiguous)
// ---------------------------------------------------------------------------
__global__ __launch_bounds__(256)
void cvt_weight_kernel(const float* __restrict__ w, _Float16* __restrict__ wh)
{
    const int i = (blockIdx.x * 256 + threadIdx.x) * 4;
    const float4 v = *(const float4*)(w + i);
    v4h o;
    o.x = (_Float16)v.x; o.y = (_Float16)v.y;
    o.z = (_Float16)v.z; o.w = (_Float16)v.w;
    *(v4h*)(wh + i) = o;
}

// ---------------------------------------------------------------------------
// Kernel 2: B-spline basis + silu -> acts fp16 (4096 x 4608), k contiguous.
// acts[b][j*512 + d] = basis_j(x[b,d]) * x[b,d]  (j<8),  silu(x)*x (j==8)
// Grid per dim is uniform linspace(-1,1,6); extended grid g[i]=g0+(i-3)h.
// ---------------------------------------------------------------------------
__global__ __launch_bounds__(256)
void kan_basis_kernel(const float* __restrict__ x, const float* __restrict__ grid,
                      _Float16* __restrict__ acts)
{
    const int idx = blockIdx.x * 256 + threadIdx.x;   // 0 .. BATCH*IN_DIM-1
    const int b = idx >> 9;          // / IN_DIM
    const int d = idx & (IN_DIM - 1);

    const float t  = x[(size_t)b * IN_DIM + d];
    const float g0 = grid[d * 6 + 0];
    const float g5 = grid[d * 6 + 5];
    const float h  = (g5 - g0) * 0.2f;          // (g5-g0)/NUM
    const float gb = g0 - 3.0f * h;             // extended grid origin, i=0

    // degree-0: indicator on [g[i], g[i+1]), i = 0..10
    float B[11];
#pragma unroll
    for (int i = 0; i < 11; ++i) {
        const float gl = gb + (float)i * h;
        B[i] = (t >= gl && t < gl + h) ? 1.0f : 0.0f;
    }

    // Cox-de Boor recurrence; uniform grid -> denominators are p*h
#pragma unroll
    for (int p = 1; p <= 3; ++p) {
        const float invph = 1.0f / ((float)p * h);
#pragma unroll
        for (int i = 0; i + p < 11; ++i) {
            const float gl    = gb + (float)i * h;
            const float left  = (t - gl) * invph;
            const float right = ((gl + (float)(p + 1) * h) - t) * invph;
            B[i] = left * B[i] + right * B[i + 1];
        }
    }

    const float silu = t * (1.0f / (1.0f + __expf(-t)));

    const size_t base = (size_t)b * KDIM + d;
#pragma unroll
    for (int j = 0; j < 8; ++j)
        acts[base + (size_t)j * IN_DIM] = (_Float16)(B[j] * t);
    acts[base + (size_t)8 * IN_DIM] = (_Float16)(silu * t);
}

// ---------------------------------------------------------------------------
// Kernel 3: C[b][o] = sum_k acts[b][k] * w[o][k] + bias[o]
// M=4096, N=512, K=4608. Block = 128 threads = 4 wave32s in a 2x2 grid.
// Per wave: 64 rows x 32 cols = 4x2 WMMA f32 16x16 accumulators.
// Double-buffered LDS (2 x (8KB A + 4KB B)), one barrier per K-step.
// ---------------------------------------------------------------------------
__global__ __launch_bounds__(128, 1)
void kan_gemm_kernel(const _Float16* __restrict__ A,   // acts, M x K
                     const _Float16* __restrict__ Bw,  // weight fp16, N x K
                     const float*    __restrict__ bias,
                     float*          __restrict__ C)
{
    __shared__ __align__(16) _Float16 As[2][BM * BK];  // 2 x 8 KB
    __shared__ __align__(16) _Float16 Bs[2][BN * BK];  // 2 x 4 KB

    const int tid   = threadIdx.x;
    const int lane  = tid & 31;
    const int wave  = tid >> 5;          // 0..3
    const int wm    = wave & 1;          // M slot: 64 rows each
    const int wn    = wave >> 1;         // N slot: 32 cols each
    const int l16   = lane & 15;
    const int lhi   = lane >> 4;         // 0 for lanes 0-15, 1 for 16-31

    const int tileM = blockIdx.x * BM;
    const int tileN = blockIdx.y * BN;

    // --- per-thread 32-bit element offsets (uniform bases advance by BK) ---
    int aoff[4], sAo[4];                 // A: 512 16B chunks / 128 threads
#pragma unroll
    for (int i = 0; i < 4; ++i) {
        const int c  = tid + i * 128;
        const int r  = c >> 2;           // 4 chunks per 64B row
        const int co = (c & 3) << 3;     // half offset within row
        aoff[i] = (tileM + r) * KDIM + co;
        sAo[i]  = r * BK + co;
    }
    int boff[2], sBo[2];                 // B: 256 16B chunks / 128 threads
#pragma unroll
    for (int i = 0; i < 2; ++i) {
        const int c  = tid + i * 128;
        const int r  = c >> 2;
        const int co = (c & 3) << 3;
        boff[i] = (tileN + r) * KDIM + co;
        sBo[i]  = r * BK + co;
    }

    v8f acc[4][2];
#pragma unroll
    for (int mt = 0; mt < 4; ++mt)
#pragma unroll
        for (int nt = 0; nt < 2; ++nt)
            acc[mt][nt] = (v8f){0.f, 0.f, 0.f, 0.f, 0.f, 0.f, 0.f, 0.f};

    // --- fragment compute from LDS buffer `cur` ---
    auto compute = [&](int cur) {
        // A 16-bit 16x32 layout: lane<16 row=l16 K{0..7,16..23}; lane>=16 K{8..15,24..31}
        v16h afrag[4], bfrag[2];
#pragma unroll
        for (int mt = 0; mt < 4; ++mt) {
            const int r  = wm * 64 + mt * 16 + l16;
            const int ko = lhi * 8;
            const v8h lo = *(const v8h*)&As[cur][r * BK + ko];
            const v8h hi = *(const v8h*)&As[cur][r * BK + 16 + ko];
            afrag[mt] = __builtin_shufflevector(lo, hi,
                0, 1, 2, 3, 4, 5, 6, 7, 8, 9, 10, 11, 12, 13, 14, 15);
        }
        // B 32x16 layout: lane = column, lanes<16 hold K{0..7,16..23}, lanes>=16 K{8..15,24..31}
#pragma unroll
        for (int nt = 0; nt < 2; ++nt) {
            const int cidx = wn * 32 + nt * 16 + l16;
            const int ko   = lhi * 8;
            const v8h lo = *(const v8h*)&Bs[cur][cidx * BK + ko];
            const v8h hi = *(const v8h*)&Bs[cur][cidx * BK + ko + 16];
            bfrag[nt] = __builtin_shufflevector(lo, hi,
                0, 1, 2, 3, 4, 5, 6, 7, 8, 9, 10, 11, 12, 13, 14, 15);
        }
#pragma unroll
        for (int mt = 0; mt < 4; ++mt)
#pragma unroll
            for (int nt = 0; nt < 2; ++nt)
                acc[mt][nt] = __builtin_amdgcn_wmma_f32_16x16x32_f16(
                    false, afrag[mt], false, bfrag[nt],
                    (short)0, acc[mt][nt], false, false);
    };

    uint4 ra[4], rb[2];
    const _Float16* pa = A;              // uniform; += BK per iteration (SALU)
    const _Float16* pb = Bw;

    // --- prologue: tile 0 -> LDS buffer 0 ---
#pragma unroll
    for (int i = 0; i < 4; ++i) ra[i] = *(const uint4*)(pa + aoff[i]);
#pragma unroll
    for (int i = 0; i < 2; ++i) rb[i] = *(const uint4*)(pb + boff[i]);
#pragma unroll
    for (int i = 0; i < 4; ++i) *(uint4*)&As[0][sAo[i]] = ra[i];
#pragma unroll
    for (int i = 0; i < 2; ++i) *(uint4*)&Bs[0][sBo[i]] = rb[i];
    __syncthreads();
    pa += BK; pb += BK;

    // --- steady state: branch-free body, last iteration peeled ---
    for (int it = 0; it < NITER - 1; ++it) {
#pragma unroll
        for (int i = 0; i < 4; ++i) ra[i] = *(const uint4*)(pa + aoff[i]);
#pragma unroll
        for (int i = 0; i < 2; ++i) rb[i] = *(const uint4*)(pb + boff[i]);
        // distance-2 speculative prefetch (OOB at tail is silently dropped)
        __builtin_prefetch(pa + BK + aoff[0], 0, 3);
        __builtin_prefetch(pb + BK + boff[0], 0, 3);
        pa += BK; pb += BK;

        compute(it & 1);

        const int nb = (it + 1) & 1;
#pragma unroll
        for (int i = 0; i < 4; ++i) *(uint4*)&As[nb][sAo[i]] = ra[i];
#pragma unroll
        for (int i = 0; i < 2; ++i) *(uint4*)&Bs[nb][sBo[i]] = rb[i];
        __syncthreads();
    }
    compute((NITER - 1) & 1);

    // --- epilogue: C/D f32 layout: vgpr i -> row i (lanes 0-15) / i+8 ---
#pragma unroll
    for (int mt = 0; mt < 4; ++mt) {
        const int rowBase = tileM + wm * 64 + mt * 16 + lhi * 8;
#pragma unroll
        for (int nt = 0; nt < 2; ++nt) {
            const int col = tileN + wn * 32 + nt * 16 + l16;
            const float bv = bias[col];
#pragma unroll
            for (int i = 0; i < 8; ++i)
                C[(size_t)(rowBase + i) * OUT_DIM + col] = acc[mt][nt][i] + bv;
        }
    }
}

// ---------------------------------------------------------------------------
extern "C" void kernel_launch(void* const* d_in, const int* in_sizes, int n_in,
                              void* d_out, int out_size, void* d_ws, size_t ws_size,
                              hipStream_t stream)
{
    const float* x      = (const float*)d_in[0];   // (4096, 512)
    const float* grid   = (const float*)d_in[1];   // (512, 6)
    const float* weight = (const float*)d_in[2];   // (512, 4608)
    const float* bias   = (const float*)d_in[3];   // (512,)
    float* out = (float*)d_out;                    // (4096, 512)

    _Float16* acts = (_Float16*)d_ws;                               // 36 MB
    _Float16* wh   = (_Float16*)((char*)d_ws +
                       (size_t)BATCH * KDIM * sizeof(_Float16));    // +4.5 MB

    // 1) weight -> fp16
    cvt_weight_kernel<<<(OUT_DIM * KDIM / 4) / 256, 256, 0, stream>>>(weight, wh);

    // 2) spline/silu activations -> fp16
    kan_basis_kernel<<<(BATCH * IN_DIM) / 256, 256, 0, stream>>>(x, grid, acts);

    // 3) WMMA GEMM + bias
    dim3 g(BATCH / BM, OUT_DIM / BN);
    kan_gemm_kernel<<<g, 128, 0, stream>>>(acts, wh, bias, out);
}
